// InteractionModule_79998060855378
// MI455X (gfx1250) — matrix operand order repaired and checked by reading, not compiled, not run on hardware
//
#include <hip/hip_runtime.h>
#include <hip/hip_bf16.h>

#define DMODEL 64
#define HEADS 4
#define DHEAD 16
#define FFDIM 2048
#define NLAYERS 3
#define SMAX 64
#define NIDS 4096

typedef __attribute__((ext_vector_type(16))) _Float16 v16h;
typedef __attribute__((ext_vector_type(8)))  float    v8f;

// ---------------- WMMA fragment helpers (wave32, 16x16x32 f16 -> f32) -------

__device__ __forceinline__ v8f wmma_f16(v16h a, v16h b, v8f c) {
    return __builtin_amdgcn_wmma_f32_16x16x32_f16(
        false, a, false, b, (short)0, c, false, false);
}

// A (16x32 f16): lane L holds row M=L&15; lanes 0-15 K=0..15, 16-31 K=16..31.
template <typename F>
__device__ __forceinline__ v16h load_a_frag(F f) {
    int L = threadIdx.x & 31;
    int m = L & 15;
    int half = (L & 16) ? 16 : 0;
    v16h a;
#pragma unroll
    for (int j = 0; j < 16; ++j) {
        int k = half + j;
        a[j] = f(m, k);
    }
    return a;
}

// B (32x16 f16): lane L holds column N=L&15; lanes 0-15 K=0..15, 16-31 K=16..31.
template <typename F>
__device__ __forceinline__ v16h load_b_frag(F f) {
    int L = threadIdx.x & 31;
    int n = L & 15;
    int kh = (L & 16) ? 16 : 0;
    v16h b;
#pragma unroll
    for (int j = 0; j < 16; ++j) b[j] = f(kh + j, n);
    return b;
}

// C/D (16x16 f32): lane L holds N=L&15; VGPR r holds M=r (+8 for lanes 16-31).
template <typename F>
__device__ __forceinline__ void store_c(const v8f& c, F f) {
    int L = threadIdx.x & 31;
    int n = L & 15;
    int mh = (L & 16) ? 8 : 0;
#pragma unroll
    for (int r = 0; r < 8; ++r) f(mh + r, n, c[r]);
}

// ---------------- helper kernels --------------------------------------------

__global__ void k_cvt_f16(const float* __restrict__ src, _Float16* __restrict__ dst,
                          int n) {
    int i = blockIdx.x * blockDim.x + threadIdx.x;
    if (i < n) dst[i] = (_Float16)src[i];
}

__global__ void k_build_groups(const int* __restrict__ ids, int* __restrict__ cnt,
                               int* __restrict__ members, int nAgents) {
    int id = blockIdx.x * blockDim.x + threadIdx.x;
    if (id >= NIDS) return;
    int c = 0;
    for (int i = 0; i < nAgents; ++i) {
        if (ids[i] == id) {
            if (c < SMAX) members[id * SMAX + c] = i;
            ++c;
        }
    }
    cnt[id] = (c > SMAX) ? SMAX : c;
}

__global__ void k_scan_rows(const int* __restrict__ cnt, int* __restrict__ outrow) {
    if (blockIdx.x == 0 && threadIdx.x == 0) {
        int r = 0;
        for (int id = 0; id < NIDS; ++id) {
            outrow[id] = (cnt[id] > 0) ? r : -1;
            if (cnt[id] > 0) ++r;
        }
    }
}

// ---------------- encoder: one workgroup per group --------------------------
// wave w owns row-tile mt=w; f16 weights streamed straight into WMMA B operands.

__global__ void __launch_bounds__(128)
k_encoder(const float* __restrict__ agents, const int* __restrict__ cnt,
          const int* __restrict__ members, const int* __restrict__ outrow,
          const _Float16* __restrict__ Wqkv16, const float* __restrict__ bqkv,
          const _Float16* __restrict__ Wo16,   const float* __restrict__ bo,
          const _Float16* __restrict__ W116,   const float* __restrict__ b1,
          const _Float16* __restrict__ W216,   const float* __restrict__ b2,
          const float* __restrict__ ln1g, const float* __restrict__ ln1b,
          const float* __restrict__ ln2g, const float* __restrict__ ln2b,
          float* __restrict__ out) {
    __shared__ float    xs[SMAX][DMODEL];            // fp32 master; rows>=S kept zero
    __shared__ _Float16 qs[SMAX][DMODEL];            // queries (pad rows harmless)
    __shared__ _Float16 ks[SMAX][DMODEL];            // keys (pad rows forced zero)
    __shared__ _Float16 vT[HEADS][DHEAD][SMAX];      // V transposed; pad cols zero
    __shared__ _Float16 attnp[HEADS][16][SMAX];      // attn probs; pad cols zero
    __shared__ _Float16 hbuf[SMAX][DMODEL];          // ctx / ffn hidden chunk (f16)
    __shared__ int      mem[SMAX];

    int gid = blockIdx.x;
    int S = cnt[gid];
    if (S <= 0) return;
    int Mtiles = (S + 15) >> 4;
    int wave = threadIdx.x >> 5;
    int L = threadIdx.x & 31;
    int ncol = L & 15;
    int mh = (L & 16) ? 8 : 0;
    int mt = wave;                    // row-tile owned by this wave
    bool act = (mt < Mtiles);         // wave-uniform

    // zero-init padded operand regions (pad regions never written after)
    {
        _Float16* vtp = &vT[0][0][0];
        _Float16* app = &attnp[0][0][0];
        for (int i = threadIdx.x; i < HEADS * DHEAD * SMAX; i += blockDim.x) {
            vtp[i] = (_Float16)0.0f;
            app[i] = (_Float16)0.0f;
        }
    }
    for (int i = threadIdx.x; i < SMAX; i += blockDim.x)
        mem[i] = (i < S) ? members[gid * SMAX + i] : 0;
    __syncthreads();
    for (int i = threadIdx.x; i < SMAX * DMODEL; i += blockDim.x) {
        int s = i >> 6, d = i & 63;
        xs[s][d] = (s < S) ? agents[mem[s] * DMODEL + d] : 0.0f;
    }
    __syncthreads();

    for (int l = 0; l < NLAYERS; ++l) {
        const _Float16* Wqkv_l = Wqkv16 + l * 3 * DMODEL * DMODEL;
        const float*    bqkv_l = bqkv + l * 3 * DMODEL;
        const _Float16* Wo_l   = Wo16 + l * DMODEL * DMODEL;
        const float*    bo_l   = bo + l * DMODEL;
        const _Float16* W1_l   = W116 + l * FFDIM * DMODEL;
        const float*    b1_l   = b1 + l * FFDIM;
        const _Float16* W2_l   = W216 + l * DMODEL * FFDIM;
        const float*    b2_l   = b2 + l * DMODEL;

        // ---------- QKV projection: [S,64] @ [64,192] -----------------------
        if (act) {
            v16h a0 = load_a_frag([&](int m, int k) -> _Float16 {
                return (_Float16)xs[mt * 16 + m][k];
            });
            v16h a1 = load_a_frag([&](int m, int k) -> _Float16 {
                return (_Float16)xs[mt * 16 + m][32 + k];
            });
            // Q tiles
#pragma unroll
            for (int nt = 0; nt < 4; ++nt) {
                v8f acc = {};
                v16h b0 = load_b_frag([&](int k, int n) -> _Float16 {
                    return Wqkv_l[(nt * 16 + n) * DMODEL + k];
                });
                acc = wmma_f16(a0, b0, acc);
                v16h b1f = load_b_frag([&](int k, int n) -> _Float16 {
                    return Wqkv_l[(nt * 16 + n) * DMODEL + 32 + k];
                });
                acc = wmma_f16(a1, b1f, acc);
                store_c(acc, [&](int m, int n, float v) {
                    int c = nt * 16 + n;
                    qs[mt * 16 + m][c] = (_Float16)(v + bqkv_l[c]);
                });
            }
            // K tiles (pad rows forced zero, branchless)
#pragma unroll
            for (int nt = 0; nt < 4; ++nt) {
                v8f acc = {};
                v16h b0 = load_b_frag([&](int k, int n) -> _Float16 {
                    return Wqkv_l[(DMODEL + nt * 16 + n) * DMODEL + k];
                });
                acc = wmma_f16(a0, b0, acc);
                v16h b1f = load_b_frag([&](int k, int n) -> _Float16 {
                    return Wqkv_l[(DMODEL + nt * 16 + n) * DMODEL + 32 + k];
                });
                acc = wmma_f16(a1, b1f, acc);
                store_c(acc, [&](int m, int n, float v) {
                    int mm = mt * 16 + m, c = nt * 16 + n;
                    float val = v + bqkv_l[DMODEL + c];
                    ks[mm][c] = (_Float16)((mm < S) ? val : 0.0f);
                });
            }
            // V tiles -> transposed store (head = nt)
#pragma unroll
            for (int nt = 0; nt < 4; ++nt) {
                v8f acc = {};
                v16h b0 = load_b_frag([&](int k, int n) -> _Float16 {
                    return Wqkv_l[(2 * DMODEL + nt * 16 + n) * DMODEL + k];
                });
                acc = wmma_f16(a0, b0, acc);
                v16h b1f = load_b_frag([&](int k, int n) -> _Float16 {
                    return Wqkv_l[(2 * DMODEL + nt * 16 + n) * DMODEL + 32 + k];
                });
                acc = wmma_f16(a1, b1f, acc);
                store_c(acc, [&](int m, int n, float v) {
                    int mm = mt * 16 + m;
                    float val = v + bqkv_l[2 * DMODEL + nt * 16 + n];
                    vT[nt][n][mm] = (_Float16)((mm < S) ? val : 0.0f);
                });
            }
        }
        __syncthreads();

        // ---------- attention: wave == head ---------------------------------
        {
            int h = wave, ho = h * DHEAD;
            for (int qt = 0; qt < Mtiles; ++qt) {
                v8f sc[4];
                {
                    v8f z = {};
#pragma unroll
                    for (int j = 0; j < 4; ++j) sc[j] = z;
                }
#pragma unroll
                for (int j = 0; j < 4; ++j) {
                    if (j < Mtiles) {
                        // K padded 16 -> 32: clamp address, select value (cndmask)
                        v16h a = load_a_frag([&](int m, int k) -> _Float16 {
                            _Float16 t0 = qs[qt * 16 + m][ho + (k & 15)];
                            return (k < DHEAD) ? t0 : (_Float16)0.0f;
                        });
                        v16h b = load_b_frag([&](int k, int n) -> _Float16 {
                            _Float16 t0 = ks[j * 16 + n][ho + (k & 15)];
                            return (k < DHEAD) ? t0 : (_Float16)0.0f;
                        });
                        sc[j] = wmma_f16(a, b, sc[j]);
                    }
                }
                // masked softmax over keys
#pragma unroll
                for (int r = 0; r < 8; ++r) {
                    float mx = -1e30f;
#pragma unroll
                    for (int j = 0; j < 4; ++j)
                        if (j < Mtiles) {
                            int key = j * 16 + ncol;
                            float v = (key < S) ? sc[j][r] * 0.25f : -1e30f;
                            sc[j][r] = v;
                            mx = fmaxf(mx, v);
                        }
                    mx = fmaxf(mx, __shfl_xor(mx, 1, 32));
                    mx = fmaxf(mx, __shfl_xor(mx, 2, 32));
                    mx = fmaxf(mx, __shfl_xor(mx, 4, 32));
                    mx = fmaxf(mx, __shfl_xor(mx, 8, 32));
                    float sum = 0.0f;
#pragma unroll
                    for (int j = 0; j < 4; ++j)
                        if (j < Mtiles) {
                            int key = j * 16 + ncol;
                            float e = (key < S) ? __expf(sc[j][r] - mx) : 0.0f;
                            sc[j][r] = e;
                            sum += e;
                        }
                    sum += __shfl_xor(sum, 1, 32);
                    sum += __shfl_xor(sum, 2, 32);
                    sum += __shfl_xor(sum, 4, 32);
                    sum += __shfl_xor(sum, 8, 32);
                    float inv = 1.0f / fmaxf(sum, 1e-20f);
#pragma unroll
                    for (int j = 0; j < 4; ++j)
                        if (j < Mtiles) sc[j][r] *= inv;
                }
                // dump probs (wave-private region; masked keys wrote exact 0)
#pragma unroll
                for (int j = 0; j < 4; ++j)
                    if (j < Mtiles) {
#pragma unroll
                        for (int r = 0; r < 8; ++r)
                            attnp[h][mh + r][j * 16 + ncol] = (_Float16)sc[j][r];
                    }
                // ctx = attn @ V_head ; K padded cols are pre-zeroed
                v8f cx = {};
#pragma unroll
                for (int kt = 0; kt < 2; ++kt) {
                    int k0 = kt * 32;
                    if (k0 < Mtiles * 16) {
                        v16h a = load_a_frag([&](int m, int k) -> _Float16 {
                            return attnp[h][m][k0 + k];
                        });
                        v16h b = load_b_frag([&](int k, int n) -> _Float16 {
                            return vT[h][n][k0 + k];
                        });
                        cx = wmma_f16(a, b, cx);
                    }
                }
                store_c(cx, [&](int m, int n, float v) {
                    hbuf[qt * 16 + m][ho + n] = (_Float16)v;
                });
            }
        }
        __syncthreads();

        // ---------- output projection + residual ----------------------------
        if (act) {
            v16h a0 = load_a_frag([&](int m, int k) -> _Float16 {
                return hbuf[mt * 16 + m][k];
            });
            v16h a1 = load_a_frag([&](int m, int k) -> _Float16 {
                return hbuf[mt * 16 + m][32 + k];
            });
#pragma unroll
            for (int nt = 0; nt < 4; ++nt) {
                v8f acc = {};
                v16h b0 = load_b_frag([&](int k, int n) -> _Float16 {
                    return Wo_l[(nt * 16 + n) * DMODEL + k];
                });
                acc = wmma_f16(a0, b0, acc);
                v16h b1f = load_b_frag([&](int k, int n) -> _Float16 {
                    return Wo_l[(nt * 16 + n) * DMODEL + 32 + k];
                });
                acc = wmma_f16(a1, b1f, acc);
                store_c(acc, [&](int m, int n, float v) {
                    int mm = mt * 16 + m, c = nt * 16 + n;
                    float nv = xs[mm][c] + v + bo_l[c];
                    xs[mm][c] = (mm < S) ? nv : 0.0f;
                });
            }
        }
        __syncthreads();
        // LayerNorm 1 (fp32)
        for (int s = threadIdx.x; s < S; s += blockDim.x) {
            float mu = 0.0f;
            for (int d = 0; d < DMODEL; ++d) mu += xs[s][d];
            mu *= (1.0f / DMODEL);
            float var = 0.0f;
            for (int d = 0; d < DMODEL; ++d) {
                float tdf = xs[s][d] - mu;
                var += tdf * tdf;
            }
            var *= (1.0f / DMODEL);
            float inv = rsqrtf(var + 1e-5f);
            for (int d = 0; d < DMODEL; ++d)
                xs[s][d] = (xs[s][d] - mu) * inv * ln1g[l * DMODEL + d] + ln1b[l * DMODEL + d];
        }
        __syncthreads();

        // ---------- FFN: 32 chunks of 64 over FF=2048 -----------------------
        v8f yacc[4];
        {
            v8f z = {};
#pragma unroll
            for (int i = 0; i < 4; ++i) yacc[i] = z;
        }
        for (int ff0 = 0; ff0 < FFDIM; ff0 += 64) {
            // h_chunk = relu(x @ W1_chunk^T + b1)
            if (act) {
                v16h a0 = load_a_frag([&](int m, int k) -> _Float16 {
                    return (_Float16)xs[mt * 16 + m][k];
                });
                v16h a1 = load_a_frag([&](int m, int k) -> _Float16 {
                    return (_Float16)xs[mt * 16 + m][32 + k];
                });
#pragma unroll
                for (int nt = 0; nt < 4; ++nt) {
                    v8f acc = {};
                    v16h b0 = load_b_frag([&](int k, int n) -> _Float16 {
                        return W1_l[(ff0 + nt * 16 + n) * DMODEL + k];
                    });
                    acc = wmma_f16(a0, b0, acc);
                    v16h b1f = load_b_frag([&](int k, int n) -> _Float16 {
                        return W1_l[(ff0 + nt * 16 + n) * DMODEL + 32 + k];
                    });
                    acc = wmma_f16(a1, b1f, acc);
                    store_c(acc, [&](int m, int n, float v) {
                        int mm = mt * 16 + m, c = nt * 16 + n;
                        float val = fmaxf(v + b1_l[ff0 + c], 0.0f);
                        hbuf[mm][c] = (_Float16)((mm < S) ? val : 0.0f);
                    });
                }
            }
            __syncthreads();
            // y += h_chunk @ W2_chunk^T  (register-resident accumulators)
            if (act) {
                v16h a0 = load_a_frag([&](int m, int k) -> _Float16 {
                    return hbuf[mt * 16 + m][k];
                });
                v16h a1 = load_a_frag([&](int m, int k) -> _Float16 {
                    return hbuf[mt * 16 + m][32 + k];
                });
#pragma unroll
                for (int nt = 0; nt < 4; ++nt) {
                    v16h b0 = load_b_frag([&](int k, int n) -> _Float16 {
                        return W2_l[(nt * 16 + n) * FFDIM + ff0 + k];
                    });
                    yacc[nt] = wmma_f16(a0, b0, yacc[nt]);
                    v16h b1f = load_b_frag([&](int k, int n) -> _Float16 {
                        return W2_l[(nt * 16 + n) * FFDIM + ff0 + 32 + k];
                    });
                    yacc[nt] = wmma_f16(a1, b1f, yacc[nt]);
                }
            }
            __syncthreads();
        }
        // FFN epilogue: residual + bias
        if (act) {
#pragma unroll
            for (int nt = 0; nt < 4; ++nt) {
                store_c(yacc[nt], [&](int m, int n, float v) {
                    int mm = mt * 16 + m, c = nt * 16 + n;
                    float nv = xs[mm][c] + v + b2_l[c];
                    xs[mm][c] = (mm < S) ? nv : 0.0f;
                });
            }
        }
        __syncthreads();
        // LayerNorm 2
        for (int s = threadIdx.x; s < S; s += blockDim.x) {
            float mu = 0.0f;
            for (int d = 0; d < DMODEL; ++d) mu += xs[s][d];
            mu *= (1.0f / DMODEL);
            float var = 0.0f;
            for (int d = 0; d < DMODEL; ++d) {
                float tdf = xs[s][d] - mu;
                var += tdf * tdf;
            }
            var *= (1.0f / DMODEL);
            float inv = rsqrtf(var + 1e-5f);
            for (int d = 0; d < DMODEL; ++d)
                xs[s][d] = (xs[s][d] - mu) * inv * ln2g[l * DMODEL + d] + ln2b[l * DMODEL + d];
        }
        __syncthreads();
    }

    // ---------- mean pool valid rows -> out[row] ----------------------------
    int row = outrow[gid];
    if (row >= 0) {
        float invS = 1.0f / (float)S;
        for (int d = threadIdx.x; d < DMODEL; d += blockDim.x) {
            float acc = 0.0f;
            for (int s = 0; s < S; ++s) acc += xs[s][d];
            out[row * DMODEL + d] = acc * invS;
        }
    }
}

// ---------------- launch ----------------------------------------------------

extern "C" void kernel_launch(void* const* d_in, const int* in_sizes, int n_in,
                              void* d_out, int out_size, void* d_ws, size_t ws_size,
                              hipStream_t stream) {
    const float* agents = (const float*)d_in[0];
    const int*   ids    = (const int*)d_in[1];
    const float* Wqkv   = (const float*)d_in[2];
    const float* bqkv   = (const float*)d_in[3];
    const float* Wo     = (const float*)d_in[4];
    const float* bo     = (const float*)d_in[5];
    const float* W1     = (const float*)d_in[6];
    const float* b1     = (const float*)d_in[7];
    const float* W2     = (const float*)d_in[8];
    const float* b2     = (const float*)d_in[9];
    const float* ln1g   = (const float*)d_in[10];
    const float* ln1b   = (const float*)d_in[11];
    const float* ln2g   = (const float*)d_in[12];
    const float* ln2b   = (const float*)d_in[13];
    float* out = (float*)d_out;
    int nAgents = in_sizes[1];

    // workspace carve
    int* cnt     = (int*)d_ws;                  // NIDS
    int* outrow  = cnt + NIDS;                  // NIDS
    int* members = outrow + NIDS;               // NIDS * SMAX
    _Float16* w16   = (_Float16*)(members + NIDS * SMAX);
    const int nWqkv = NLAYERS * 3 * DMODEL * DMODEL;   // 36864
    const int nWo   = NLAYERS * DMODEL * DMODEL;       // 12288
    const int nW1   = NLAYERS * FFDIM * DMODEL;        // 393216
    const int nW2   = NLAYERS * DMODEL * FFDIM;        // 393216
    _Float16* wqkv16 = w16;
    _Float16* wo16   = wqkv16 + nWqkv;
    _Float16* w116   = wo16 + nWo;
    _Float16* w216   = w116 + nW1;

    // one-time (per call) weight downconversion fp32 -> f16
    hipLaunchKernelGGL(k_cvt_f16, dim3((nWqkv + 255) / 256), dim3(256), 0, stream,
                       Wqkv, wqkv16, nWqkv);
    hipLaunchKernelGGL(k_cvt_f16, dim3((nWo + 255) / 256), dim3(256), 0, stream,
                       Wo, wo16, nWo);
    hipLaunchKernelGGL(k_cvt_f16, dim3((nW1 + 255) / 256), dim3(256), 0, stream,
                       W1, w116, nW1);
    hipLaunchKernelGGL(k_cvt_f16, dim3((nW2 + 255) / 256), dim3(256), 0, stream,
                       W2, w216, nW2);

    hipLaunchKernelGGL(k_build_groups, dim3(NIDS / 256), dim3(256), 0, stream,
                       ids, cnt, members, nAgents);
    hipLaunchKernelGGL(k_scan_rows, dim3(1), dim3(1), 0, stream, cnt, outrow);
    hipLaunchKernelGGL(k_encoder, dim3(NIDS), dim3(128), 0, stream,
                       agents, cnt, members, outrow,
                       wqkv16, bqkv, wo16, bo, w116, b1, w216, b2,
                       ln1g, ln1b, ln2g, ln2b, out);
    (void)n_in; (void)out_size; (void)ws_size;
}
